// VectorQuantizerEMA_49838800502811
// MI455X (gfx1250) — compile-verified, matrix operand order checked by reading, not run on hardware
//
#include <hip/hip_runtime.h>
#include <hip/hip_bf16.h>

#define NROWS  32768
#define DDIM   256
#define KCODES 1024

typedef __attribute__((ext_vector_type(16))) __bf16 v16bf;
typedef __attribute__((ext_vector_type(8)))  __bf16 v8bf;
typedef __attribute__((ext_vector_type(8)))  float  v8f;
typedef __attribute__((ext_vector_type(4)))  float  v4f;
typedef __attribute__((ext_vector_type(4)))  unsigned v4u;
typedef __attribute__((ext_vector_type(8)))  int    v8i;
typedef __attribute__((ext_vector_type(4)))  int    v4i;

// ---------------------------------------------------------------------------
// k0: per-code squared norms + zero counts + pre-split embedding to bf16
//     hi/lo, row layout [code][ hi 0..255 | lo 256..511 ]  (one TDM tile/row)
// ---------------------------------------------------------------------------
__global__ void k_enorm(const float* __restrict__ emb,
                        float* __restrict__ enorm,
                        unsigned* __restrict__ counts,
                        __bf16* __restrict__ ebG)
{
    __shared__ float red[256];
    const int k = blockIdx.x;
    const int t = threadIdx.x;
    const float v = emb[(size_t)k * DDIM + t];
    const __bf16 h = (__bf16)v;
    ebG[(size_t)k * 512 + t]       = h;
    ebG[(size_t)k * 512 + 256 + t] = (__bf16)(v - (float)h);
    red[t] = v * v;
    __syncthreads();
    for (int s = 128; s > 0; s >>= 1) {
        if (t < s) red[t] += red[t + s];
        __syncthreads();
    }
    if (t == 0) { enorm[k] = red[0]; counts[k] = 0u; }
}

// ---------------------------------------------------------------------------
// TDM: DMA one 64-code chunk (64 rows x 1024B) of the pre-split bf16
// embedding into LDS, padding 16B after every 1024B row (pad_interval=7:
// 256 DWORDs, pad_amount=3: 4 DWORDs) -> LDS row stride 1040B, which keeps
// the later ds_load_b128 fragment reads bank-conflict-free per 16-lane half.
// D# layout per CDNA5 ISA.
// ---------------------------------------------------------------------------
__device__ __forceinline__ void tdm_load_chunk(const __bf16* ebG, int chunk,
                                               unsigned lds_addr)
{
    const unsigned long long ga =
        (unsigned long long)(const void*)(ebG + (size_t)chunk * 64 * 512);
    v4u g0;
    g0[0] = 1u;                                   // count=1 (valid descriptor)
    g0[1] = lds_addr;                             // LDS byte address
    g0[2] = (unsigned)(ga & 0xffffffffull);       // global_addr[31:0]
    g0[3] = (unsigned)((ga >> 32) & 0x1ffffffull) // global_addr[56:32]
          | (2u << 30);                           // type=2 ("image")
    v8i g1;
    g1[0] = (1 << 16)   // data_size = 2 bytes
          | (1 << 20)   // pad_enable
          | (7 << 22)   // pad_interval: 256 DWORDs (1024B)
          | (3 << 25);  // pad_amount:   4 DWORDs (16B)
    g1[1] = (int)(512u << 16);  // tensor_dim0 = 512 elements (low16)
    g1[2] = (int)(64u  << 16);  // tensor_dim1 = 64 rows (low16)
    g1[3] = (int)(512u << 16);  // tile_dim0 = 512 elements
    g1[4] = 64;                 // tile_dim1 = 64 rows, tile_dim2 = 0
    g1[5] = 512;                // tensor_dim0_stride = 512 elements
    g1[6] = 0;
    g1[7] = 0;
    v4i g2 = {0, 0, 0, 0};      // 2-D tile: dims 2..4 unused
    v4i g3 = {0, 0, 0, 0};
    asm volatile("tensor_load_to_lds %0, %1, %2, %3"
                 :
                 : "s"(g0), "s"(g1), "s"(g2), "s"(g3)
                 : "memory");
}

// ---------------------------------------------------------------------------
// k1: distance GEMM via bf16x3 WMMA + argmin, TDM double-buffered B tiles.
//   score_k = ||e_k||^2 - 2 * z.e_k
// One WG = 8 waves = 128 rows; 16 chunks of 64 codes; wave 0 drives the TDM.
// ---------------------------------------------------------------------------
__global__ void
__launch_bounds__(256)
k_argmin_wmma(const float* __restrict__ z,
              const __bf16* __restrict__ ebG,
              const float* __restrict__ enorm,
              int*   __restrict__ idx_out,
              float* __restrict__ idx_out_f)
{
    constexpr int RS = 520;                       // bf16 row stride (1040 B)
    __shared__ __bf16 sbuf[2][64 * RS];           // 2 x 66.5 KB

    const int tid  = threadIdx.x;
    const int lane = tid & 31;
    const int wave = tid >> 5;
    const int half = lane >> 4;                   // K-half select (A/B frag)
    const int l16  = lane & 15;
    const int rowBase = blockIdx.x * 128 + wave * 16;
    const int r = rowBase + l16;

    // ---- prefetch chunk 0 while we build A fragments ----
    if (wave == 0)
        tdm_load_chunk(ebG, 0, (unsigned)(size_t)(void*)&sbuf[0][0]);

    // ---- load A fragments (hi/lo bf16 split) into registers, once ----
    v16bf aH[8], aL[8];
    const v4f* zr4 = (const v4f*)(z + (size_t)r * DDIM);
    #pragma unroll
    for (int ks = 0; ks < 8; ++ks) {
        const int b4 = ks * 8 + half * 2;         // float4 index of K-chunk
        v4f c0 = zr4[b4 + 0];
        v4f c1 = zr4[b4 + 1];
        v4f c2 = zr4[b4 + 4];                     // +16 dims
        v4f c3 = zr4[b4 + 5];
        float f[16] = { c0[0],c0[1],c0[2],c0[3], c1[0],c1[1],c1[2],c1[3],
                        c2[0],c2[1],c2[2],c2[3], c3[0],c3[1],c3[2],c3[3] };
        #pragma unroll
        for (int i = 0; i < 16; ++i) {
            __bf16 h = (__bf16)f[i];
            aH[ks][i] = h;
            aL[ks][i] = (__bf16)(f[i] - (float)h);
        }
    }

    float bestS[8];
    int   bestI[8];
    #pragma unroll
    for (int v = 0; v < 8; ++v) { bestS[v] = 3.0e38f; bestI[v] = 0x7fffffff; }

    for (int c = 0; c < 16; ++c) {
        if (wave == 0) __builtin_amdgcn_s_wait_tensorcnt(0);  // chunk c landed
        __syncthreads();                          // publish to all waves; also
                                                  // fences readers of buf[(c+1)&1]
        if (wave == 0 && c + 1 < 16)              // overlap DMA with compute
            tdm_load_chunk(ebG, c + 1,
                           (unsigned)(size_t)(void*)&sbuf[(c + 1) & 1][0]);

        const __bf16* bufc = sbuf[c & 1];
        for (int nt = 0; nt < 4; ++nt) {
            const int codeLocal = nt * 16 + l16;
            const int code      = c * 64 + codeLocal;
            v8f acc = {};
            #pragma unroll
            for (int ks = 0; ks < 8; ++ks) {
                const int boff = codeLocal * RS + ks * 32 + half * 8;
                v8bf bh0 = *(const v8bf*)&bufc[boff];
                v8bf bh1 = *(const v8bf*)&bufc[boff + 16];
                v8bf bl0 = *(const v8bf*)&bufc[boff + 256];
                v8bf bl1 = *(const v8bf*)&bufc[boff + 272];
                v16bf bh, bl;
                #pragma unroll
                for (int i = 0; i < 8; ++i) {
                    bh[i] = bh0[i]; bh[i + 8] = bh1[i];
                    bl[i] = bl0[i]; bl[i + 8] = bl1[i];
                }
                acc = __builtin_amdgcn_wmma_f32_16x16x32_bf16(
                          false, aH[ks], false, bh, (short)0, acc, false, false);
                acc = __builtin_amdgcn_wmma_f32_16x16x32_bf16(
                          false, aH[ks], false, bl, (short)0, acc, false, false);
                acc = __builtin_amdgcn_wmma_f32_16x16x32_bf16(
                          false, aL[ks], false, bh, (short)0, acc, false, false);
            }
            const float en = enorm[code];
            #pragma unroll
            for (int v = 0; v < 8; ++v) {
                const float s = en - 2.0f * acc[v];
                if (s < bestS[v] || (s == bestS[v] && code < bestI[v])) {
                    bestS[v] = s; bestI[v] = code;
                }
            }
        }
    }

    // ---- argmin across the 16 N-lanes of each half-group ----
    #pragma unroll
    for (int off = 1; off < 16; off <<= 1) {
        #pragma unroll
        for (int v = 0; v < 8; ++v) {
            const float os = __shfl_xor(bestS[v], off, 32);
            const int   oi = __shfl_xor(bestI[v], off, 32);
            if (os < bestS[v] || (os == bestS[v] && oi < bestI[v])) {
                bestS[v] = os; bestI[v] = oi;
            }
        }
    }
    if (l16 == 0) {
        #pragma unroll
        for (int v = 0; v < 8; ++v) {
            const int row = rowBase + v + 8 * half;  // C layout: M = v + 8*half
            idx_out[row]   = bestI[v];
            idx_out_f[row] = (float)bestI[v];
        }
    }
}

// ---------------------------------------------------------------------------
// k2: gather z_q, per-row loss partial (fixed shuffle tree), integer counts
// ---------------------------------------------------------------------------
__global__ void k_gather(const float* __restrict__ z,
                         const float* __restrict__ emb,
                         const int*   __restrict__ idx,
                         float* __restrict__ out_zq,
                         float* __restrict__ loss_part,
                         unsigned* __restrict__ counts)
{
    const int lane = threadIdx.x & 31;
    const int row  = blockIdx.x * 8 + (threadIdx.x >> 5);
    const int k    = idx[row];
    const v4f* er4 = (const v4f*)(emb + (size_t)k * DDIM);
    const v4f* zr4 = (const v4f*)(z + (size_t)row * DDIM);
    v4f*       qr4 = (v4f*)(out_zq + (size_t)row * DDIM);
    float s = 0.0f;
    #pragma unroll
    for (int j = 0; j < 2; ++j) {
        const int d4 = lane + 32 * j;
        v4f q  = er4[d4];
        v4f zz = zr4[d4];
        qr4[d4] = q;
        #pragma unroll
        for (int q4 = 0; q4 < 4; ++q4) {
            const float dd = q[q4] - zz[q4];
            s += dd * dd;
        }
    }
    #pragma unroll
    for (int off = 16; off > 0; off >>= 1) s += __shfl_xor(s, off, 32);
    if (lane == 0) {
        loss_part[row] = s;
        atomicAdd(&counts[k], 1u);               // integer: deterministic
    }
}

// ---------------------------------------------------------------------------
// k3: deterministic segment-sum partials (thread t owns dim column t)
// ---------------------------------------------------------------------------
#define CR 64
__global__ void
__launch_bounds__(256)
k_embed_partial(const float* __restrict__ z,
                const int*   __restrict__ idx,
                float* __restrict__ partial)
{
    __shared__ float acc[CR * DDIM];             // 64 KB
    const int t       = threadIdx.x;
    const int chunkId = blockIdx.x & 15;         // 16 n-chunks
    const int range   = blockIdx.x >> 4;         // 16 code ranges
    const int kbase   = range * CR;
    for (int j = t; j < CR * DDIM; j += 256) acc[j] = 0.0f;
    __syncthreads();
    const int n0 = chunkId * 2048;
    for (int n = n0; n < n0 + 2048; ++n) {
        const int k = idx[n];
        if ((unsigned)(k - kbase) < (unsigned)CR) {
            acc[(k - kbase) * DDIM + t] += z[(size_t)n * DDIM + t];
        }
    }
    __syncthreads();
    float* dst = partial + ((size_t)chunkId * KCODES + kbase) * DDIM;
    for (int j = t; j < CR * DDIM; j += 256) dst[j] = acc[j];
}

// ---------------------------------------------------------------------------
// k4a (single block): loss scalar, new_cluster_size, n = sum(new_cluster_size)
// ---------------------------------------------------------------------------
__global__ void k_finalize_a(const float* __restrict__ csz,
                             const unsigned* __restrict__ counts,
                             const float* __restrict__ loss_part,
                             float* __restrict__ out_loss,
                             float* __restrict__ out_ncs,
                             float* __restrict__ ws_n)
{
    __shared__ float red[256];
    const int t = threadIdx.x;
    float s = 0.0f;
    for (int j = 0; j < 128; ++j) s += loss_part[t + 256 * j];
    red[t] = s; __syncthreads();
    for (int st = 128; st > 0; st >>= 1) {
        if (t < st) red[t] += red[t + st];
        __syncthreads();
    }
    if (t == 0) out_loss[0] = 0.25f * red[0] / 8388608.0f;
    __syncthreads();
    float nsum = 0.0f;
    for (int j = 0; j < 4; ++j) {
        const int k = t + 256 * j;
        const float v = csz[k] * 0.99f + 0.01f * (float)counts[k];
        out_ncs[k] = v;
        nsum += v;
    }
    red[t] = nsum; __syncthreads();
    for (int st = 128; st > 0; st >>= 1) {
        if (t < st) red[t] += red[t + st];
        __syncthreads();
    }
    if (t == 0) ws_n[0] = red[0];
}

// ---------------------------------------------------------------------------
// k4b: reduce chunk partials (fixed order), EMA avg, normalized embedding
// ---------------------------------------------------------------------------
__global__ void k_finalize_b(const float* __restrict__ eavg,
                             const float* __restrict__ partial,
                             const float* __restrict__ out_ncs,
                             const float* __restrict__ ws_n,
                             float* __restrict__ out_navg,
                             float* __restrict__ out_nemb)
{
    const int k = blockIdx.x;
    const int d = threadIdx.x;
    float s = 0.0f;
    #pragma unroll
    for (int ch = 0; ch < 16; ++ch)
        s += partial[((size_t)ch * KCODES + k) * DDIM + d];
    const float navg = eavg[(size_t)k * DDIM + d] * 0.99f + 0.01f * s;
    out_navg[(size_t)k * DDIM + d] = navg;
    const float cs = (out_ncs[k] + 1e-6f) / (ws_n[0] + (float)KCODES * 1e-6f);
    out_nemb[(size_t)k * DDIM + d] = navg / cs;
}

// ---------------------------------------------------------------------------
extern "C" void kernel_launch(void* const* d_in, const int* in_sizes, int n_in,
                              void* d_out, int out_size, void* d_ws, size_t ws_size,
                              hipStream_t stream)
{
    const float* z    = (const float*)d_in[0];   // [32768, 256]
    const float* emb  = (const float*)d_in[1];   // [1024, 256]
    const float* csz  = (const float*)d_in[2];   // [1024]
    const float* eavg = (const float*)d_in[3];   // [1024, 256]

    float* out      = (float*)d_out;             // tuple, concatenated
    float* out_zq   = out;                       // 8388608
    float* out_loss = out + 8388608;             // 1
    float* out_idx  = out + 8388609;             // 32768
    float* out_ncs  = out + 8421377;             // 1024
    float* out_navg = out + 8422401;             // 262144
    float* out_nemb = out + 8684545;             // 262144

    char* ws = (char*)d_ws;
    float*    enorm   = (float*)(ws + 0);            // 4 KB
    int*      idxbuf  = (int*)(ws + 4096);           // 128 KB
    unsigned* counts  = (unsigned*)(ws + 135168);    // 4 KB
    float*    lpart   = (float*)(ws + 139264);       // 128 KB
    float*    ws_n    = (float*)(ws + 270336);       // 4 B
    __bf16*   ebG     = (__bf16*)(ws + (2 << 20));   // 1 MB: [1024][512] bf16
    float*    partial = (float*)(ws + (4 << 20));    // 16 MB: [16][1024][256]

    k_enorm<<<KCODES, 256, 0, stream>>>(emb, enorm, counts, ebG);
    k_argmin_wmma<<<NROWS / 128, 256, 0, stream>>>(z, ebG, enorm, idxbuf, out_idx);
    k_gather<<<NROWS / 8, 256, 0, stream>>>(z, emb, idxbuf, out_zq, lpart, counts);
    k_embed_partial<<<256, 256, 0, stream>>>(z, idxbuf, partial);
    k_finalize_a<<<1, 256, 0, stream>>>(csz, counts, lpart, out_loss, out_ncs, ws_n);
    k_finalize_b<<<KCODES, 256, 0, stream>>>(eavg, partial, out_ncs, ws_n,
                                             out_navg, out_nemb);
}
